// RoPECrossAttention_13245679140912
// MI455X (gfx1250) — matrix operand order repaired and checked by reading, not compiled, not run on hardware
//
#include <hip/hip_runtime.h>

typedef __attribute__((ext_vector_type(16))) _Float16 v16h;
typedef __attribute__((ext_vector_type(8)))  _Float16 v8h;
typedef __attribute__((ext_vector_type(8)))  float    v8f;
typedef __attribute__((ext_vector_type(4)))  int      v4i;

union AFrag { v16h v; v8h h8[2]; };
union BFrag { v16h v; v8h h8[2]; };

// --------------------------------------------------------------------------
// CDNA5 async global->LDS copy support (ASYNCcnt path), with safe fallback.
// Probe result: builtin exists; param0 is v4i in AS1 (printed as __device__).
// --------------------------------------------------------------------------
#if defined(__gfx1250__) && __has_builtin(__builtin_amdgcn_global_load_async_to_lds_b128)
#define USE_ASYNC_LDS 1
#else
#define USE_ASYNC_LDS 0
#endif

#if USE_ASYNC_LDS
static __device__ inline void async_copy_b128(const _Float16* gsrc, _Float16* ldst) {
    __builtin_amdgcn_global_load_async_to_lds_b128(
        (__attribute__((address_space(1))) v4i*)gsrc,
        (__attribute__((address_space(3))) v4i*)ldst,
        0, 0);
}
static __device__ inline void wait_async_all() {
#if __has_builtin(__builtin_amdgcn_s_wait_asynccnt)
    __builtin_amdgcn_s_wait_asynccnt(0);
#else
    asm volatile("s_wait_asynccnt 0x0" ::: "memory");
#endif
}
#endif

static __device__ inline v8f vzero8f() {
    v8f z;
#pragma unroll
    for (int i = 0; i < 8; ++i) z[i] = 0.f;
    return z;
}

// ---------------------------------------------------------------------------
// Kernel 1: row LayerNorm (D=1024) -> f16
// ---------------------------------------------------------------------------
__global__ __launch_bounds__(256) void ln_to_f16_kernel(
    const float* __restrict__ x, _Float16* __restrict__ y)
{
    __shared__ float rs[256];
    __shared__ float rs2[256];
    const int row = blockIdx.x;
    const float* xr = x + (size_t)row * 1024;
    float v[4], s = 0.f, s2 = 0.f;
#pragma unroll
    for (int i = 0; i < 4; ++i) {
        v[i] = xr[threadIdx.x + 256 * i];
        s += v[i];
        s2 += v[i] * v[i];
    }
    rs[threadIdx.x] = s;
    rs2[threadIdx.x] = s2;
    __syncthreads();
    for (int off = 128; off > 0; off >>= 1) {
        if (threadIdx.x < off) {
            rs[threadIdx.x]  += rs[threadIdx.x + off];
            rs2[threadIdx.x] += rs2[threadIdx.x + off];
        }
        __syncthreads();
    }
    const float mu  = rs[0] * (1.f / 1024.f);
    const float var = rs2[0] * (1.f / 1024.f) - mu * mu;
    const float inv = rsqrtf(var + 1e-5f);
    _Float16* yr = y + (size_t)row * 1024;
#pragma unroll
    for (int i = 0; i < 4; ++i)
        yr[threadIdx.x + 256 * i] = (_Float16)((v[i] - mu) * inv);
}

// ---------------------------------------------------------------------------
// Kernel 2: W'[i][j] = f16(scale[i] * W[i][j])   (scale==nullptr -> 1.0)
// ---------------------------------------------------------------------------
__global__ __launch_bounds__(256) void weight_prep_kernel(
    const float* __restrict__ W, const float* __restrict__ scale,
    _Float16* __restrict__ Wh)
{
    const int idx = blockIdx.x * 256 + threadIdx.x;   // 0 .. 1024*1024-1
    const float s = scale ? scale[idx >> 10] : 1.f;
    Wh[idx] = (_Float16)(s * W[idx]);
}

// ---------------------------------------------------------------------------
// Kernel 3: bias'[j] = bias[j] + sum_i affine_bias[i] * W[i][j]
// ---------------------------------------------------------------------------
__global__ __launch_bounds__(256) void bias_fold_kernel(
    const float* __restrict__ W, const float* __restrict__ ab,
    const float* __restrict__ b, float* __restrict__ bout)
{
    const int j = blockIdx.x * 256 + threadIdx.x;     // 0..1023
    float acc = b[j];
    for (int i = 0; i < 1024; ++i)
        acc += ab[i] * W[(size_t)i * 1024 + j];
    bout[j] = acc;
}

// ---------------------------------------------------------------------------
// Kernel 4: tiled f16 WMMA GEMM   C[M x 1024] = A[M x 1024] * B[1024 x 1024]
//   MODE 0: out f16, scattered into [B,H,N,hd] layout, + bias
//   MODE 1: out f32 row-major, + bias
//   block tile 128x64, K step 32, 8 waves in a 4x2 grid of 32x32 wave tiles
// ---------------------------------------------------------------------------
template <int MODE>
__global__ __launch_bounds__(256) void gemm_f16_kernel(
    const _Float16* __restrict__ A, const _Float16* __restrict__ Bw,
    const float* __restrict__ bias, void* __restrict__ Out)
{
    __shared__ _Float16 As[128][48];   // 96B row stride (16B aligned)
    __shared__ _Float16 Bs[64][48];    // transposed: Bs[n][k]

    const int bm0  = blockIdx.x * 128;
    const int bn0  = blockIdx.y * 64;
    const int tid  = threadIdx.x;
    const int lane = tid & 31;
    const int wave = tid >> 5;     // 0..7
    const int wm   = wave >> 1;    // 0..3
    const int wn   = wave & 1;     // 0..1
    const int r    = lane & 15;
    const int hi   = lane >> 4;

    v8f c[2][2];
#pragma unroll
    for (int mt = 0; mt < 2; ++mt)
#pragma unroll
        for (int ct = 0; ct < 2; ++ct) c[mt][ct] = vzero8f();

    for (int k0 = 0; k0 < 1024; k0 += 32) {
        // A tile: 128x32 halves, 16 halves/thread as 2x b128
#pragma unroll
        for (int c8 = 0; c8 < 2; ++c8) {
            const int q    = tid * 2 + c8;      // 0..511
            const int arow = q >> 2;            // 0..127
            const int acol = (q & 3) * 8;
            const _Float16* gsrc = A + (size_t)(bm0 + arow) * 1024 + k0 + acol;
#if USE_ASYNC_LDS
            async_copy_b128(gsrc, &As[arow][acol]);   // VGPR-free, ASYNCcnt
#else
            *(v8h*)&As[arow][acol] = *(const v8h*)gsrc;
#endif
        }
        // B tile transposed: 32x64 -> Bs[n][k]  (register transpose: sync path)
        {
            const int krow = tid >> 3;          // 0..31
            const int ncol = (tid & 7) * 8;
            v8h tv = *(const v8h*)(Bw + (size_t)(k0 + krow) * 1024 + bn0 + ncol);
#pragma unroll
            for (int e = 0; e < 8; ++e) Bs[ncol + e][krow] = tv[e];
        }
        if (k0 + 32 < 1024)
            __builtin_prefetch(A + (size_t)(bm0 + (tid >> 1)) * 1024 + k0 + 32, 0, 0);
#if USE_ASYNC_LDS
        wait_async_all();
#endif
        __syncthreads();

        AFrag a[2];
        BFrag bfr[2];
#pragma unroll
        for (int mt = 0; mt < 2; ++mt) {
            const int mrow = wm * 32 + mt * 16 + r;
            a[mt].h8[0] = *(const v8h*)&As[mrow][8 * hi];        // K 0..7 / 8..15
            a[mt].h8[1] = *(const v8h*)&As[mrow][16 + 8 * hi];   // K 16..23 / 24..31
        }
#pragma unroll
        for (int ct = 0; ct < 2; ++ct) {
            const int ncol = wn * 32 + ct * 16 + r;
            bfr[ct].h8[0] = *(const v8h*)&Bs[ncol][8 * hi];      // K 0..7  / 8..15
            bfr[ct].h8[1] = *(const v8h*)&Bs[ncol][16 + 8 * hi]; // K 16..23 / 24..31
        }
#pragma unroll
        for (int mt = 0; mt < 2; ++mt)
#pragma unroll
            for (int ct = 0; ct < 2; ++ct)
                c[mt][ct] = __builtin_amdgcn_wmma_f32_16x16x32_f16(
                    false, a[mt].v, false, bfr[ct].v, (short)0, c[mt][ct],
                    false, false);
        __syncthreads();
    }

    // epilogue
#pragma unroll
    for (int mt = 0; mt < 2; ++mt) {
#pragma unroll
        for (int ct = 0; ct < 2; ++ct) {
            const int col = bn0 + wn * 32 + ct * 16 + r;
            const float bb = bias ? bias[col] : 0.f;
#pragma unroll
            for (int v = 0; v < 8; ++v) {
                const int grow = bm0 + wm * 32 + mt * 16 + hi * 8 + v;
                const float val = c[mt][ct][v] + bb;
                if (MODE == 0) {
                    const int batch = grow >> 10, n = grow & 1023;
                    const int h = col >> 6, d = col & 63;
                    _Float16* dst = (_Float16*)Out;
                    dst[(((size_t)(batch * 16 + h) * 1024 + n) << 6) + d] =
                        (_Float16)val;
                } else {
                    float* dst = (float*)Out;
                    dst[(size_t)grow * 1024 + col] = val;
                }
            }
        }
    }
}

// ---------------------------------------------------------------------------
// Kernel 5: RoPE in-place on f16 [B,H,N,64]; one thread per rotate-half pair
// ---------------------------------------------------------------------------
__global__ __launch_bounds__(256) void rope_kernel(
    _Float16* __restrict__ X, const float* __restrict__ cs,
    const float* __restrict__ sn)
{
    const int idx = blockIdx.x * 256 + threadIdx.x;  // B*H*N*32 total
    const int d = idx & 31;
    const int n = (idx >> 5) & 1023;
    const size_t bh = (size_t)(idx >> 15);
    _Float16* p = X + (bh << 16) + ((size_t)n << 6);
    const float x0 = (float)p[d], x1 = (float)p[d + 32];
    const float c0 = cs[n * 64 + d],      s0 = sn[n * 64 + d];
    const float c1 = cs[n * 64 + d + 32], s1 = sn[n * 64 + d + 32];
    p[d]      = (_Float16)(x0 * c0 - x1 * s0);   // rot = -x[d+32]
    p[d + 32] = (_Float16)(x1 * c1 + x0 * s1);   // rot =  x[d-32]
}

// ---------------------------------------------------------------------------
// Kernel 6: flash attention, one block (4 waves) per (b,h, 64 query rows).
// Each wave owns 16 query rows; online softmax stays within half-waves.
// ---------------------------------------------------------------------------
__global__ __launch_bounds__(128) void flash_attn_kernel(
    const _Float16* __restrict__ Q, const _Float16* __restrict__ K,
    const _Float16* __restrict__ V, _Float16* __restrict__ O)
{
    __shared__ _Float16 Ks[64][72];        // [key][hd]
    __shared__ _Float16 Vt[64][72];        // [hd][key]
    __shared__ _Float16 Ps[4][16][72];     // per-wave P staging

    const int tid  = threadIdx.x;
    const int lane = tid & 31;
    const int wave = tid >> 5;
    const int r    = lane & 15;
    const int hi   = lane >> 4;
    const int qblk = blockIdx.x & 15;
    const int bh   = blockIdx.x >> 4;      // b*16 + h
    const int b    = bh >> 4, h = bh & 15;

    const _Float16* Qbh = Q + ((size_t)bh << 16);
    const _Float16* Kbh = K + ((size_t)bh << 16);
    const _Float16* Vbh = V + ((size_t)bh << 16);

    // Q fragments for this wave's 16 rows, held for the whole kernel
    const int qrow = qblk * 64 + wave * 16 + r;
    AFrag aq[2];
#pragma unroll
    for (int kc = 0; kc < 2; ++kc) {
        const _Float16* src = Qbh + (size_t)qrow * 64 + kc * 32;
        aq[kc].h8[0] = *(const v8h*)(src + 8 * hi);
        aq[kc].h8[1] = *(const v8h*)(src + 16 + 8 * hi);
    }

    float mrow[8], lrow[8];
    v8f acc[4];
#pragma unroll
    for (int v = 0; v < 8; ++v) { mrow[v] = -1e30f; lrow[v] = 0.f; }
#pragma unroll
    for (int ct = 0; ct < 4; ++ct) acc[ct] = vzero8f();

    for (int kt = 0; kt < 16; ++kt) {
        const int kbase = kt * 64;
        // cooperative K tile (async, straight to LDS) + transposed V tile
#pragma unroll
        for (int c8 = 0; c8 < 4; ++c8) {
            const int q   = tid * 4 + c8;   // 0..511
            const int row = q >> 3;
            const int col = (q & 7) * 8;
            const _Float16* ksrc = Kbh + (size_t)(kbase + row) * 64 + col;
#if USE_ASYNC_LDS
            async_copy_b128(ksrc, &Ks[row][col]);
#else
            *(v8h*)&Ks[row][col] = *(const v8h*)ksrc;
#endif
            v8h tv = *(const v8h*)(Vbh + (size_t)(kbase + row) * 64 + col);
#pragma unroll
            for (int e = 0; e < 8; ++e) Vt[col + e][row] = tv[e];
        }
#if USE_ASYNC_LDS
        wait_async_all();
#endif
        __syncthreads();

        // S = Q K^T * 1/sqrt(64)
        float p[4][8];
        float rowmax[8];
#pragma unroll
        for (int v = 0; v < 8; ++v) rowmax[v] = -1e30f;
#pragma unroll
        for (int nt = 0; nt < 4; ++nt) {
            v8f s = vzero8f();
#pragma unroll
            for (int kc = 0; kc < 2; ++kc) {
                BFrag bk;   // B[kdim][key]: lane's key = nt*16 + r
                bk.h8[0] = *(const v8h*)&Ks[nt * 16 + r][kc * 32 + 8 * hi];
                bk.h8[1] = *(const v8h*)&Ks[nt * 16 + r][kc * 32 + 16 + 8 * hi];
                s = __builtin_amdgcn_wmma_f32_16x16x32_f16(
                    false, aq[kc].v, false, bk.v, (short)0, s, false, false);
            }
#pragma unroll
            for (int v = 0; v < 8; ++v) {
                p[nt][v] = s[v] * 0.125f;
                rowmax[v] = fmaxf(rowmax[v], p[nt][v]);
            }
        }
        // row max across the 16 lanes sharing this half-wave
#pragma unroll
        for (int v = 0; v < 8; ++v) {
            float m = rowmax[v];
#pragma unroll
            for (int mk = 1; mk < 16; mk <<= 1)
                m = fmaxf(m, __shfl_xor(m, mk, 32));
            rowmax[v] = m;
        }
        float corr[8], rsum[8];
#pragma unroll
        for (int v = 0; v < 8; ++v) {
            const float mn = fmaxf(mrow[v], rowmax[v]);
            corr[v] = __expf(mrow[v] - mn);
            mrow[v] = mn;
            rsum[v] = 0.f;
        }
#pragma unroll
        for (int nt = 0; nt < 4; ++nt)
#pragma unroll
            for (int v = 0; v < 8; ++v) {
                const float e = __expf(p[nt][v] - mrow[v]);
                rsum[v] += e;
                Ps[wave][hi * 8 + v][nt * 16 + r] = (_Float16)e;
            }
#pragma unroll
        for (int v = 0; v < 8; ++v) {
            float sm = rsum[v];
#pragma unroll
            for (int mk = 1; mk < 16; mk <<= 1)
                sm += __shfl_xor(sm, mk, 32);
            lrow[v] = lrow[v] * corr[v] + sm;
        }
#pragma unroll
        for (int ct = 0; ct < 4; ++ct)
#pragma unroll
            for (int v = 0; v < 8; ++v) acc[ct][v] *= corr[v];

        __syncthreads();   // P visible to whole wave, all tiles loaded

        // acc += P @ V
        const _Float16* prow = &Ps[wave][r][0];
#pragma unroll
        for (int ct = 0; ct < 4; ++ct) {
#pragma unroll
            for (int kc = 0; kc < 2; ++kc) {
                AFrag ap;
                ap.h8[0] = *(const v8h*)(prow + kc * 32 + 8 * hi);
                ap.h8[1] = *(const v8h*)(prow + kc * 32 + 16 + 8 * hi);
                BFrag bv;
                bv.h8[0] = *(const v8h*)&Vt[ct * 16 + r][kc * 32 + 8 * hi];
                bv.h8[1] = *(const v8h*)&Vt[ct * 16 + r][kc * 32 + 16 + 8 * hi];
                acc[ct] = __builtin_amdgcn_wmma_f32_16x16x32_f16(
                    false, ap.v, false, bv.v, (short)0, acc[ct], false, false);
            }
        }
        __syncthreads();   // before next tile overwrites Ks/Vt/Ps
    }

    // finalize: out f16 in [B,N,D] layout for the O-projection GEMM
#pragma unroll
    for (int ct = 0; ct < 4; ++ct) {
        const int d = ct * 16 + r;
#pragma unroll
        for (int v = 0; v < 8; ++v) {
            const int row = qblk * 64 + wave * 16 + hi * 8 + v;
            const float val = acc[ct][v] / lrow[v];
            O[((size_t)b * 1024 + row) * 1024 + h * 64 + d] = (_Float16)val;
        }
    }
}

// ---------------------------------------------------------------------------
// Host launcher
// ---------------------------------------------------------------------------
extern "C" void kernel_launch(void* const* d_in, const int* in_sizes, int n_in,
                              void* d_out, int out_size, void* d_ws, size_t ws_size,
                              hipStream_t stream)
{
    (void)in_sizes; (void)n_in; (void)out_size; (void)ws_size;
    const float* q_in   = (const float*)d_in[0];
    const float* kv_in  = (const float*)d_in[1];
    const float* q_cos  = (const float*)d_in[2];
    const float* q_sin  = (const float*)d_in[3];
    const float* kv_cos = (const float*)d_in[4];
    const float* kv_sin = (const float*)d_in[5];
    const float* aq_s   = (const float*)d_in[6];
    const float* aq_b   = (const float*)d_in[7];
    const float* ak_s   = (const float*)d_in[8];
    const float* ak_b   = (const float*)d_in[9];
    const float* av_s   = (const float*)d_in[10];
    const float* av_b   = (const float*)d_in[11];
    const float* Wq     = (const float*)d_in[12];
    const float* Wk     = (const float*)d_in[13];
    const float* Wv     = (const float*)d_in[14];
    const float* Wo     = (const float*)d_in[15];
    const float* bq     = (const float*)d_in[16];
    const float* bk     = (const float*)d_in[17];
    const float* bv     = (const float*)d_in[18];
    const float* bo     = (const float*)d_in[19];

    char* ws = (char*)d_ws;
    size_t off = 0;
    auto carve = [&](size_t bytes) -> void* {
        void* p = ws + off;
        off += (bytes + 255) & ~(size_t)255;
        return p;
    };
    const size_t ACT = (size_t)8192 * 1024 * sizeof(_Float16);   // 16 MB
    const size_t WT  = (size_t)1024 * 1024 * sizeof(_Float16);   // 2 MB
    _Float16* Xq   = (_Float16*)carve(ACT);
    _Float16* Xkv  = (_Float16*)carve(ACT);
    _Float16* WqH  = (_Float16*)carve(WT);
    _Float16* WkH  = (_Float16*)carve(WT);
    _Float16* WvH  = (_Float16*)carve(WT);
    _Float16* WoH  = (_Float16*)carve(WT);
    float*    bqF  = (float*)carve(1024 * sizeof(float));
    float*    bkF  = (float*)carve(1024 * sizeof(float));
    float*    bvF  = (float*)carve(1024 * sizeof(float));
    _Float16* Qf   = (_Float16*)carve(ACT);
    _Float16* Kf   = (_Float16*)carve(ACT);
    _Float16* Vf   = (_Float16*)carve(ACT);
    _Float16* AttO = (_Float16*)carve(ACT);

    // 1) layernorm -> f16
    ln_to_f16_kernel<<<8192, 256, 0, stream>>>(q_in, Xq);
    ln_to_f16_kernel<<<8192, 256, 0, stream>>>(kv_in, Xkv);

    // 2) fold affine scale into f16 weights; fold affine bias into GEMM bias
    weight_prep_kernel<<<4096, 256, 0, stream>>>(Wq, aq_s, WqH);
    weight_prep_kernel<<<4096, 256, 0, stream>>>(Wk, ak_s, WkH);
    weight_prep_kernel<<<4096, 256, 0, stream>>>(Wv, av_s, WvH);
    weight_prep_kernel<<<4096, 256, 0, stream>>>(Wo, nullptr, WoH);
    bias_fold_kernel<<<4, 256, 0, stream>>>(Wq, aq_b, bq, bqF);
    bias_fold_kernel<<<4, 256, 0, stream>>>(Wk, ak_b, bk, bkF);
    bias_fold_kernel<<<4, 256, 0, stream>>>(Wv, av_b, bv, bvF);

    // 3) QKV projections (WMMA), writing [B,H,N,hd] f16
    dim3 ggrid(8192 / 128, 1024 / 64);
    gemm_f16_kernel<0><<<ggrid, 256, 0, stream>>>(Xq,  WqH, bqF, Qf);
    gemm_f16_kernel<0><<<ggrid, 256, 0, stream>>>(Xkv, WkH, bkF, Kf);
    gemm_f16_kernel<0><<<ggrid, 256, 0, stream>>>(Xkv, WvH, bvF, Vf);

    // 4) RoPE on Q and K
    rope_kernel<<<16384, 256, 0, stream>>>(Qf, q_cos, q_sin);
    rope_kernel<<<16384, 256, 0, stream>>>(Kf, kv_cos, kv_sin);

    // 5) flash attention -> [B,N,D] f16
    flash_attn_kernel<<<2048, 128, 0, stream>>>(Qf, Kf, Vf, AttO);

    // 6) output projection (WMMA) -> f32 d_out
    gemm_f16_kernel<1><<<ggrid, 256, 0, stream>>>(AttO, WoH, bo, (float*)d_out);
}
